// DrBC_79293686219296
// MI455X (gfx1250) — compile-verified, matrix operand order checked by reading, not run on hardware
//
#include <hip/hip_runtime.h>
#include <hip/hip_bf16.h>
#include <math.h>

typedef __bf16 bf16_t;
typedef bf16_t v16bf __attribute__((ext_vector_type(16)));
typedef float  v8f   __attribute__((ext_vector_type(8)));

#define PDIM 128
#define G3P  384   // 3*P
#define QDIM 64

// ---------------------------------------------------------------------------
// small utility kernels
// ---------------------------------------------------------------------------
__global__ void fill_f32_kernel(float* p, float v, int n) {
    int i = blockIdx.x * blockDim.x + threadIdx.x;
    if (i < n) p[i] = v;
}

__global__ void deg_accum_kernel(const int* __restrict__ col, float* deg, int n_edges) {
    int i = blockIdx.x * blockDim.x + threadIdx.x;
    if (i < n_edges) atomicAdd(&deg[col[i]], 1.0f);
}

__global__ void rsqrt_kernel(float* p, int n) {
    int i = blockIdx.x * blockDim.x + threadIdx.x;
    if (i < n) p[i] = rsqrtf(p[i]);
}

__global__ void edge_norm_kernel(const int* __restrict__ row, const int* __restrict__ col,
                                 const float* __restrict__ dinv, float* __restrict__ nrm,
                                 int n_edges) {
    int i = blockIdx.x * blockDim.x + threadIdx.x;
    if (i < n_edges) nrm[i] = dinv[row[i]] * dinv[col[i]];
}

__global__ void cvt_bf16_kernel(const float* __restrict__ src, bf16_t* __restrict__ dst, int n) {
    int i = blockIdx.x * blockDim.x + threadIdx.x;
    if (i < n) dst[i] = (bf16_t)src[i];
}

// ---------------------------------------------------------------------------
// input layer: h = l2norm(relu(x @ w0^T + b0)); also initializes zmax = h
// one wave per node, 4 features per lane
// ---------------------------------------------------------------------------
__global__ __launch_bounds__(256) void input_layer_kernel(
    const float* __restrict__ x, const float* __restrict__ w0, const float* __restrict__ b0,
    float* __restrict__ h, float* __restrict__ zmax, int n_nodes)
{
    int node = (int)((blockIdx.x * blockDim.x + threadIdx.x) >> 5);
    int lane = threadIdx.x & 31;
    if (node >= n_nodes) return;
    float x0 = x[node * 3 + 0], x1 = x[node * 3 + 1], x2 = x[node * 3 + 2];
    float vals[4];
    float ss = 0.0f;
#pragma unroll
    for (int i = 0; i < 4; ++i) {
        int p = lane + 32 * i;
        float v = fmaf(x0, w0[p * 3 + 0], fmaf(x1, w0[p * 3 + 1], fmaf(x2, w0[p * 3 + 2], b0[p])));
        v = fmaxf(v, 0.0f);
        vals[i] = v;
        ss += v * v;
    }
#pragma unroll
    for (int off = 16; off; off >>= 1) ss += __shfl_xor(ss, off, 32);
    float scale = 1.0f / fmaxf(sqrtf(ss), 1e-12f);
#pragma unroll
    for (int i = 0; i < 4; ++i) {
        int p = lane + 32 * i;
        float v = vals[i] * scale;
        h[node * PDIM + p] = v;
        zmax[node * PDIM + p] = v;
    }
}

// ---------------------------------------------------------------------------
// edge scatter: aggr[col] += norm * h[row]   (one wave per edge, coalesced 512B)
// ---------------------------------------------------------------------------
__global__ __launch_bounds__(256) void scatter_kernel(
    const int* __restrict__ row, const int* __restrict__ col,
    const float* __restrict__ nrm, const float* __restrict__ h,
    float* __restrict__ aggr, int n_edges)
{
    int e = (int)((blockIdx.x * blockDim.x + threadIdx.x) >> 5);
    int lane = threadIdx.x & 31;
    if (e >= n_edges) return;
    int r = row[e], c = col[e];
    float nm = nrm[e];
#pragma unroll
    for (int i = 0; i < 4; ++i) {
        int f = lane + 32 * i;
        atomicAdd(&aggr[(long)c * PDIM + f], nm * h[(long)r * PDIM + f]);
    }
}

// ---------------------------------------------------------------------------
// GRU layer with WMMA (bf16 -> f32):
//   gi = h @ w_ih^T, gh = aggr @ w_hh^T   (each [16 nodes, 384])
//   gates + l2norm, in-place h update, zmax running max
// block = 256 threads (8 waves), 16 nodes per block
// ---------------------------------------------------------------------------
__global__ __launch_bounds__(256) void gru_wmma_kernel(
    float* h,                             // [N,128] in/out (in-place per-row safe)
    const float* __restrict__ aggr,       // [N,128]
    const bf16_t* __restrict__ wih,       // [384,128] bf16 row-major
    const bf16_t* __restrict__ whh,       // [384,128] bf16 row-major
    const float* __restrict__ bih,        // [384]
    const float* __restrict__ bhh,        // [384]
    float* __restrict__ zmax,             // [N,128]
    int n_nodes)
{
    __shared__ bf16_t hA[16][PDIM];       // 4 KB
    __shared__ bf16_t aA[16][PDIM];       // 4 KB
    __shared__ float  gi[16][G3P];        // 24 KB
    __shared__ float  gh[16][G3P];        // 24 KB
    __shared__ float  scale_s[16];

    const int tid = threadIdx.x;
    const int lane = tid & 31;
    const int wid = tid >> 5;
    const int nodeBase = blockIdx.x * 16;

    // ---- stage h / aggr tiles to LDS as bf16 ----
    for (int idx = tid; idx < 16 * PDIM; idx += 256) {
        int m = idx >> 7, j = idx & (PDIM - 1);
        int node = nodeBase + m;
        float hv = 0.0f, av = 0.0f;
        if (node < n_nodes) {
            hv = h[(long)node * PDIM + j];
            av = aggr[(long)node * PDIM + j];
        }
        hA[m][j] = (bf16_t)hv;
        aA[m][j] = (bf16_t)av;
    }
    __syncthreads();

    // ---- 48 WMMA output tiles: t in [0,24) -> gi, [24,48) -> gh ----
    const int mrow = lane & 15;   // M (A) / N (B) index held by this lane
    const int hi = lane >> 4;     // lane half selects K sub-block per ISA layout
    for (int t = wid; t < 48; t += 8) {           // wave-uniform loop: EXEC stays all-ones
        const bool isGi = (t < 24);
        const int jBase = (isGi ? t : t - 24) * 16;
        const bf16_t* W = isGi ? wih : whh;
        const bf16_t(*A)[PDIM] = isGi ? hA : aA;
        v8f acc = {};
        for (int kb = 0; kb < PDIM; kb += 32) {
            v16bf a, b;
            // 16-bit A 16x32 layout: lanes 0-15 hold K{0-7,16-23}, lanes 16-31 K{8-15,24-31}
#pragma unroll
            for (int i = 0; i < 8; ++i) {
                int k0 = kb + hi * 8 + i;
                int k1 = kb + 16 + hi * 8 + i;
                a[i]     = A[mrow][k0];
                a[i + 8] = A[mrow][k1];
                // B[k][n] = W[jBase+n][k]; symmetric K striping, lane holds column n=mrow
                b[i]     = W[(jBase + mrow) * PDIM + k0];
                b[i + 8] = W[(jBase + mrow) * PDIM + k1];
            }
            acc = __builtin_amdgcn_wmma_f32_16x16x32_bf16(
                false, a, false, b, (short)0, acc, false, false);
        }
        // C/D 16x16 f32 layout: VGPR v, lanes 0-15 -> M=v, lanes 16-31 -> M=8+v; N=lane&15
        float* G = isGi ? &gi[0][0] : &gh[0][0];
        const int mOut = hi * 8;
        const int nOut = lane & 15;
#pragma unroll
        for (int v = 0; v < 8; ++v)
            G[(mOut + v) * G3P + jBase + nOut] = acc[v];
    }
    __syncthreads();

    // ---- fused GRU gate math; stash h' back into gi[:, 0:128] ----
    for (int idx = tid; idx < 16 * PDIM; idx += 256) {
        int m = idx >> 7, j = idx & (PDIM - 1);
        int node = nodeBase + m;
        float gir = gi[m][j]            + bih[j];
        float ghr = gh[m][j]            + bhh[j];
        float giz = gi[m][PDIM + j]     + bih[PDIM + j];
        float ghz = gh[m][PDIM + j]     + bhh[PDIM + j];
        float gin = gi[m][2 * PDIM + j] + bih[2 * PDIM + j];
        float ghn = gh[m][2 * PDIM + j] + bhh[2 * PDIM + j];
        float r = 1.0f / (1.0f + __expf(-(gir + ghr)));
        float z = 1.0f / (1.0f + __expf(-(giz + ghz)));
        float nn = tanhf(gin + r * ghn);
        float hid = (node < n_nodes) ? aggr[(long)node * PDIM + j] : 0.0f;
        gi[m][j] = (1.0f - z) * nn + z * hid;
    }
    __syncthreads();

    // ---- per-node L2 norm ----
    if (tid < 16) {
        float ss = 0.0f;
        for (int j = 0; j < PDIM; ++j) { float v = gi[tid][j]; ss += v * v; }
        scale_s[tid] = 1.0f / fmaxf(sqrtf(ss), 1e-12f);
    }
    __syncthreads();

    for (int idx = tid; idx < 16 * PDIM; idx += 256) {
        int m = idx >> 7, j = idx & (PDIM - 1);
        int node = nodeBase + m;
        if (node < n_nodes) {
            float hv = gi[m][j] * scale_s[m];
            h[(long)node * PDIM + j] = hv;
            float* zm = &zmax[(long)node * PDIM + j];
            *zm = fmaxf(*zm, hv);
        }
    }
}

// ---------------------------------------------------------------------------
// decoder: out = relu(z @ w4^T + b4) @ w5^T + b5  (one wave per node)
// ---------------------------------------------------------------------------
__global__ __launch_bounds__(256) void decoder_kernel(
    const float* __restrict__ z, const float* __restrict__ w4, const float* __restrict__ b4,
    const float* __restrict__ w5, const float* __restrict__ b5,
    float* __restrict__ out, int n_nodes)
{
    int node = (int)((blockIdx.x * blockDim.x + threadIdx.x) >> 5);
    int lane = threadIdx.x & 31;
    if (node >= n_nodes) return;
    float partial = 0.0f;
#pragma unroll
    for (int s = 0; s < 2; ++s) {
        int q = lane + 32 * s;
        float acc = b4[q];
        for (int k = 0; k < PDIM; ++k)
            acc = fmaf(z[(long)node * PDIM + k], w4[q * PDIM + k], acc);
        partial += fmaxf(acc, 0.0f) * w5[q];
    }
#pragma unroll
    for (int off = 16; off; off >>= 1) partial += __shfl_xor(partial, off, 32);
    if (lane == 0) out[node] = partial + b5[0];
}

// ---------------------------------------------------------------------------
// launch
// ---------------------------------------------------------------------------
static inline size_t align256(size_t x) { return (x + 255) & ~(size_t)255; }

extern "C" void kernel_launch(void* const* d_in, const int* in_sizes, int n_in,
                              void* d_out, int out_size, void* d_ws, size_t ws_size,
                              hipStream_t stream) {
    const float* x     = (const float*)d_in[0];
    const int*   ei    = (const int*)d_in[1];      // [2, E]
    const float* w0_w  = (const float*)d_in[2];
    const float* w0_b  = (const float*)d_in[3];
    const float* wih_f = (const float*)d_in[4];
    const float* whh_f = (const float*)d_in[5];
    const float* bih   = (const float*)d_in[6];
    const float* bhh   = (const float*)d_in[7];
    const float* w4_w  = (const float*)d_in[8];
    const float* w4_b  = (const float*)d_in[9];
    const float* w5_w  = (const float*)d_in[10];
    const float* w5_b  = (const float*)d_in[11];
    float* out = (float*)d_out;

    const int N = in_sizes[0] / 3;
    const int E = in_sizes[1] / 2;
    const int* row = ei;
    const int* col = ei + E;

    // workspace layout
    char* ws = (char*)d_ws;
    size_t off = 0;
    float* deg  = (float*)(ws + off); off = align256(off + (size_t)N * sizeof(float));
    float* nrm  = (float*)(ws + off); off = align256(off + (size_t)E * sizeof(float));
    float* h    = (float*)(ws + off); off = align256(off + (size_t)N * PDIM * sizeof(float));
    float* aggr = (float*)(ws + off); off = align256(off + (size_t)N * PDIM * sizeof(float));
    float* zmax = (float*)(ws + off); off = align256(off + (size_t)N * PDIM * sizeof(float));
    bf16_t* wih = (bf16_t*)(ws + off); off = align256(off + (size_t)G3P * PDIM * sizeof(bf16_t));
    bf16_t* whh = (bf16_t*)(ws + off); off = align256(off + (size_t)G3P * PDIM * sizeof(bf16_t));
    (void)ws_size; (void)n_in; (void)out_size;

    const int T = 256;

    // 1) degrees & per-edge norm
    fill_f32_kernel<<<(N + T - 1) / T, T, 0, stream>>>(deg, 1.0f, N);
    deg_accum_kernel<<<(E + T - 1) / T, T, 0, stream>>>(col, deg, E);
    rsqrt_kernel<<<(N + T - 1) / T, T, 0, stream>>>(deg, N);
    edge_norm_kernel<<<(E + T - 1) / T, T, 0, stream>>>(row, col, deg, nrm, E);

    // 2) weights -> bf16 (stays L2-resident, halves WMMA B-fragment traffic)
    cvt_bf16_kernel<<<(G3P * PDIM + T - 1) / T, T, 0, stream>>>(wih_f, wih, G3P * PDIM);
    cvt_bf16_kernel<<<(G3P * PDIM + T - 1) / T, T, 0, stream>>>(whh_f, whh, G3P * PDIM);

    // 3) input layer (also inits zmax)
    input_layer_kernel<<<((size_t)N * 32 + T - 1) / T, T, 0, stream>>>(x, w0_w, w0_b, h, zmax, N);

    // 4) message-passing layers
    for (int layer = 0; layer < 4; ++layer) {
        hipMemsetAsync(aggr, 0, (size_t)N * PDIM * sizeof(float), stream);
        scatter_kernel<<<((size_t)E * 32 + T - 1) / T, T, 0, stream>>>(row, col, nrm, h, aggr, E);
        gru_wmma_kernel<<<(N + 15) / 16, T, 0, stream>>>(h, aggr, wih, whh, bih, bhh, zmax, N);
    }

    // 5) decoder
    decoder_kernel<<<((size_t)N * 32 + T - 1) / T, T, 0, stream>>>(zmax, w4_w, w4_b, w5_w, w5_b, out, N);
}
